// FocalTransformerBlock_60344290509102
// MI455X (gfx1250) — matrix operand order repaired and checked by reading, not compiled, not run on hardware
//
#include <hip/hip_runtime.h>
#include <hip/hip_bf16.h>
#include <math.h>

typedef __attribute__((ext_vector_type(16))) __bf16 v16bf;
typedef __attribute__((ext_vector_type(8)))  __bf16 v8bf;
typedef __attribute__((ext_vector_type(8)))  float  v8f;
typedef __attribute__((ext_vector_type(4)))  int    v4i;

#define DEV __device__ __forceinline__

static constexpr int Bb = 4, Lm = 4096, Cc = 512, Hh = 8, HDd = 64;
static constexpr int WSQ = 64, WSk = 16, Sdim = 768, nWq = 64;
static constexpr int RPBH = 256;
static constexpr float QSCALE = 0.125f;          // HD^-0.5
static constexpr int LVL_ROWS = 7168;            // 4096+2048+1024 rows per batch

// -------- CDNA5 async global->LDS path (guarded; sync fallback) ----------
#if defined(__HIP_DEVICE_COMPILE__) && defined(__has_builtin)
#if __has_builtin(__builtin_amdgcn_global_load_async_to_lds_b128) && \
    __has_builtin(__builtin_amdgcn_s_wait_asynccnt)
#define USE_ASYNC_LDS 1
#endif
#endif

#ifdef USE_ASYNC_LDS
typedef __attribute__((address_space(1))) v4i* gas_v4i;
typedef __attribute__((address_space(3))) v4i* las_v4i;
#endif

DEV void lds_cp8(__bf16* l, const __bf16* g) {
#ifdef USE_ASYNC_LDS
  __builtin_amdgcn_global_load_async_to_lds_b128((gas_v4i)(g), (las_v4i)(l), 0, 0);
#else
  *(v8bf*)l = *(const v8bf*)g;
#endif
}
DEV void async_wait() {
#ifdef USE_ASYNC_LDS
  __builtin_amdgcn_s_wait_asynccnt(0);
#endif
}

// ---------------------------------------------------------------- helpers
DEV v16bf make_frag(const __bf16* p0, const __bf16* p1) {
  v8bf lo = *(const v8bf*)p0;
  v8bf hi = *(const v8bf*)p1;
  v16bf r;
#pragma unroll
  for (int i = 0; i < 8; ++i) { r[i] = lo[i]; r[i + 8] = hi[i]; }
  return r;
}

// j-th closest key-window index for a query centered at cq (stable argsort
// of |cq - ck|: at equal distance the smaller index (cq-d) comes first).
DEV int topk_kw(int cq, int nk, int j) {
  int cnt = 0;
  for (int d = 0; d < nk; ++d) {
    int lo = cq - d;
    if (lo >= 0)            { if (cnt == j) return lo; ++cnt; }
    int hi = cq + d;
    if (d > 0 && hi < nk)   { if (cnt == j) return hi; ++cnt; }
  }
  return 0;
}

// ---------------------------------------------------------------- small kernels
__global__ void f32_to_bf16_kernel(const float* __restrict__ src,
                                   __bf16* __restrict__ dst, int n) {
  int i = blockIdx.x * blockDim.x + threadIdx.x;
  if (i < n) dst[i] = (__bf16)src[i];
}

// LayerNorm over C=512, one wave per row, output bf16
__global__ __launch_bounds__(256)
void ln_bf16_kernel(const float* __restrict__ x, const float* __restrict__ sc,
                    const float* __restrict__ bi, __bf16* __restrict__ out, int rows) {
  int wv = threadIdx.x >> 5, lane = threadIdx.x & 31;
  int row = blockIdx.x * 8 + wv;
  if (row >= rows) return;
  const float* xr = x + (size_t)row * Cc;
  float vals[16], sum = 0.f, sq = 0.f;
#pragma unroll
  for (int i = 0; i < 16; ++i) {
    float v = xr[lane + 32 * i];
    vals[i] = v; sum += v; sq += v * v;
  }
#pragma unroll
  for (int m = 1; m < 32; m <<= 1) {
    sum += __shfl_xor(sum, m, 32);
    sq  += __shfl_xor(sq,  m, 32);
  }
  float mean = sum * (1.f / 512.f);
  float var  = sq  * (1.f / 512.f) - mean * mean;
  float rs   = rsqrtf(var + 1e-5f);
  __bf16* orow = out + (size_t)row * Cc;
#pragma unroll
  for (int i = 0; i < 16; ++i) {
    int c = lane + 32 * i;
    orow[c] = (__bf16)((vals[i] - mean) * rs * sc[c] + bi[c]);
  }
}

// mean-pool rows by factor f: out rows = Bb*(Lm/f)
__global__ void pool_kernel(const __bf16* __restrict__ hb, __bf16* __restrict__ out,
                            int f, int rowsOut) {
  int idx = blockIdx.x * blockDim.x + threadIdx.x;
  int total = rowsOut * Cc;
  if (idx >= total) return;
  int r = idx / Cc, c = idx - r * Cc;
  int Lf = Lm / f;
  int b = r / Lf, t = r - b * Lf;
  const __bf16* base = hb + ((size_t)(b * Lm + t * f)) * Cc + c;
  float acc = 0.f;
  for (int i = 0; i < f; ++i) acc += (float)base[(size_t)i * Cc];
  out[idx] = (__bf16)(acc / (float)f);
}

// gathered key-row index table: gidx[w][s] = lvl_off + kw*16 + (s%16)
__global__ void gidx_kernel(int* __restrict__ gidx) {
  int idx = blockIdx.x * blockDim.x + threadIdx.x;
  if (idx >= nWq * Sdim) return;
  int w = idx / Sdim, s = idx - w * Sdim;
  int l = s >> 8, rem = s & 255, j = rem >> 4, within = rem & 15;
  int nk = 256 >> l;
  int fq = nk / nWq;
  int cq = w * fq + fq / 2;
  int lvlOff = (l == 0) ? 0 : ((l == 1) ? 4096 : 6144);
  int kw = topk_kw(cq, nk, j);
  gidx[idx] = lvlOff + kw * WSk + within;
}

// relative-position-bias MLP: bias[w][q][s][h]  (batch independent)
__global__ __launch_bounds__(256)
void rpb_kernel(const float* __restrict__ w1, const float* __restrict__ b1,
                const float* __restrict__ w2, const float* __restrict__ b2,
                __bf16* __restrict__ biasb) {
  __shared__ float sw1[RPBH], sb1[RPBH], sw2[RPBH * 8], sb2[8];
  int w = blockIdx.x, q = blockIdx.y;
  for (int i = threadIdx.x; i < RPBH; i += blockDim.x) { sw1[i] = w1[i]; sb1[i] = b1[i]; }
  for (int i = threadIdx.x; i < RPBH * 8; i += blockDim.x) sw2[i] = w2[i];
  if (threadIdx.x < 8) sb2[threadIdx.x] = b2[threadIdx.x];
  __syncthreads();
  for (int s = threadIdx.x; s < Sdim; s += blockDim.x) {
    int l = s >> 8, rem = s & 255, j = rem >> 4, sin = rem & 15;
    int nk = 256 >> l, fq = nk / nWq;
    int cq = w * fq + fq / 2;
    int f = 1 << l;
    int kw = topk_kw(cq, nk, j);
    float coord = (float)(cq - kw);
    float rel = (coord * (float)(WSk * f) + (float)(q - WSQ / 2)
                 - (float)((sin - WSk / 2) * f)) / (float)Lm;
    float acc[8];
#pragma unroll
    for (int h = 0; h < 8; ++h) acc[h] = sb2[h];
    for (int i = 0; i < RPBH; ++i) {
      float hv = rel * sw1[i] + sb1[i];
      if (hv > 0.f) {
#pragma unroll
        for (int h = 0; h < 8; ++h) acc[h] += hv * sw2[i * 8 + h];
      }
    }
    size_t base = (((size_t)w * WSQ + q) * Sdim + s) * Hh;
#pragma unroll
    for (int h = 0; h < 8; ++h) biasb[base + h] = (__bf16)acc[h];
  }
}

// ---------------------------------------------------------------- WMMA GEMM
// block tile 128(M) x 128(N), 8 waves as 4(M) x 2(N), wave tile 32x64,
// double-buffered LDS, async global->LDS staging for the A tile.
enum { MODE_Q = 0, MODE_KV = 1, MODE_PROJ = 2, MODE_MLP1 = 3, MODE_MLP2 = 4 };

__global__ __launch_bounds__(256)
void gemm_wmma_kernel(const __bf16* __restrict__ A, int lda,
                      const __bf16* __restrict__ W, int N, int K,
                      const float* __restrict__ bias, int mode,
                      __bf16* __restrict__ outb, float* __restrict__ outf,
                      const float* __restrict__ resid,
                      __bf16* __restrict__ outb2, int Ll, int lvlOff) {
  __shared__ __bf16 aT[2][128][40];   // [buf][m][k]
  __shared__ __bf16 bT[2][128][40];   // [buf][n][k] (transposed)
  int tid = threadIdx.x;
  int wv = tid >> 5, lane = tid & 31;
  int l16 = lane & 15, lhi = lane >> 4;
  int wvm = wv & 3, wvn = wv >> 2;          // 4 x 2 wave grid
  int nTile = blockIdx.x * 128;
  int mTile = blockIdx.y * 128;

  v8f acc[2][4];
#pragma unroll
  for (int mf = 0; mf < 2; ++mf)
#pragma unroll
    for (int nf = 0; nf < 4; ++nf)
#pragma unroll
      for (int j = 0; j < 8; ++j) acc[mf][nf][j] = 0.f;

  int arow = tid >> 1, ach = (tid & 1) * 16;   // A: 128 rows x 32 (16 elems/thread)
  int bk = tid >> 3, bn = (tid & 7) * 16;      // B: 32 rows x 128 (16 elems/thread)

  const __bf16* aSrc = A + (size_t)(mTile + arow) * lda + ach;
  const __bf16* bSrc = W + (size_t)bk * N + nTile + bn;

  // prologue: stage k-tile 0 into buffer 0
  lds_cp8(&aT[0][arow][ach], aSrc);
  lds_cp8(&aT[0][arow][ach + 8], aSrc + 8);
  {
    v8bf w0 = *(const v8bf*)bSrc;
    v8bf w1 = *(const v8bf*)(bSrc + 8);
#pragma unroll
    for (int j = 0; j < 8; ++j) { bT[0][bn + j][bk] = w0[j]; bT[0][bn + 8 + j][bk] = w1[j]; }
  }
  async_wait();
  __syncthreads();

  int KT = K >> 5;
  for (int kt = 0; kt < KT; ++kt) {
    int cur = kt & 1, nxt = cur ^ 1;
    bool more = (kt + 1) < KT;
    v8bf wn0, wn1;
    if (more) {   // issue next-tile loads before computing (overlap with WMMA)
      const __bf16* as = aSrc + (size_t)(kt + 1) * 32;
      lds_cp8(&aT[nxt][arow][ach], as);
      lds_cp8(&aT[nxt][arow][ach + 8], as + 8);
      const __bf16* bs = bSrc + (size_t)(kt + 1) * 32 * N;
      wn0 = *(const v8bf*)bs;
      wn1 = *(const v8bf*)(bs + 8);
      if (kt + 2 < KT) {
        __builtin_prefetch(as + 32, 0, 1);
        __builtin_prefetch(bs + (size_t)32 * N, 0, 1);
      }
    }
    // batched fragment loads, then 8 back-to-back WMMAs
    v16bf af0 = make_frag(&aT[cur][wvm * 32 + l16][lhi * 8],
                          &aT[cur][wvm * 32 + l16][16 + lhi * 8]);
    v16bf af1 = make_frag(&aT[cur][wvm * 32 + 16 + l16][lhi * 8],
                          &aT[cur][wvm * 32 + 16 + l16][16 + lhi * 8]);
    v16bf bf[4];
#pragma unroll
    for (int nf = 0; nf < 4; ++nf)
      bf[nf] = make_frag(&bT[cur][wvn * 64 + nf * 16 + l16][lhi * 8],
                         &bT[cur][wvn * 64 + nf * 16 + l16][16 + lhi * 8]);
#pragma unroll
    for (int nf = 0; nf < 4; ++nf)
      acc[0][nf] = __builtin_amdgcn_wmma_f32_16x16x32_bf16(
          false, af0, false, bf[nf], (short)0, acc[0][nf], false, false);
#pragma unroll
    for (int nf = 0; nf < 4; ++nf)
      acc[1][nf] = __builtin_amdgcn_wmma_f32_16x16x32_bf16(
          false, af1, false, bf[nf], (short)0, acc[1][nf], false, false);
    if (more) {
#pragma unroll
      for (int j = 0; j < 8; ++j) { bT[nxt][bn + j][bk] = wn0[j]; bT[nxt][bn + 8 + j][bk] = wn1[j]; }
      async_wait();
    }
    __syncthreads();
  }

  // epilogue
#pragma unroll
  for (int mf = 0; mf < 2; ++mf) {
#pragma unroll
    for (int nf = 0; nf < 4; ++nf) {
      int col = nTile + wvn * 64 + nf * 16 + l16;
      float bv = bias ? bias[col] : 0.f;
#pragma unroll
      for (int r = 0; r < 8; ++r) {
        int row = mTile + wvm * 32 + mf * 16 + lhi * 8 + r;
        float v = acc[mf][nf][r] + bv;
        if (mode == MODE_Q) {
          outb[(size_t)row * N + col] = (__bf16)(v * QSCALE);
        } else if (mode == MODE_KV) {
          int b = row / Ll, t = row - b * Ll;
          size_t orow = (size_t)b * LVL_ROWS + lvlOff + t;
          if (col < 512) outb [orow * 512 + col]         = (__bf16)v;
          else           outb2[orow * 512 + (col - 512)] = (__bf16)v;
        } else if (mode == MODE_PROJ) {
          size_t o = (size_t)row * 512 + col;
          outf[o] = v + resid[o];
        } else if (mode == MODE_MLP1) {
          float u = v;
          float g = 0.5f * u * (1.f + tanhf(0.7978845608f * (u + 0.044715f * u * u * u)));
          outb[(size_t)row * N + col] = (__bf16)g;
        } else {  // MODE_MLP2
          size_t o = (size_t)row * 512 + col;
          outf[o] = v + resid[o];
        }
      }
    }
  }
}

// ---------------------------------------------------------------- flash attention
__global__ __launch_bounds__(128)
void flash_attn_kernel(const __bf16* __restrict__ qb, const __bf16* __restrict__ kb,
                       const __bf16* __restrict__ vb, const __bf16* __restrict__ biasb,
                       const int* __restrict__ gidx, __bf16* __restrict__ ao) {
  __shared__ __bf16 qT[64][72];        // [q][d]
  __shared__ __bf16 kT[64][72];        // [s][d]
  __shared__ __bf16 vT[64][72];        // transposed: [d][s]
  __shared__ __bf16 pT[4][16][72];     // per-wave P tile [q][s]

  int h = blockIdx.x, w = blockIdx.y, b = blockIdx.z;
  int tid = threadIdx.x;
  int wv = tid >> 5, lane = tid & 31;
  int l16 = lane & 15, lhi = lane >> 4;
  int m0 = wv * 16;

  {   // stage q tile (64 rows x 64 dims for this head) via async copies
    int row = tid >> 1, ch = (tid & 1) * 32;
    const __bf16* src = qb + ((size_t)(b * Lm + w * WSQ + row)) * Cc + h * HDd + ch;
#pragma unroll
    for (int j = 0; j < 4; ++j) lds_cp8(&qT[row][ch + j * 8], src + j * 8);
  }

  v8f o[4];
#pragma unroll
  for (int i = 0; i < 4; ++i)
#pragma unroll
    for (int j = 0; j < 8; ++j) o[i][j] = 0.f;
  float m_run[8], l_run[8];
#pragma unroll
  for (int r = 0; r < 8; ++r) { m_run[r] = -1e30f; l_run[r] = 0.f; }

  const int* gw = gidx + (size_t)w * Sdim;

  for (int c = 0; c < Sdim / 64; ++c) {
    int s0 = c * 64;
    __syncthreads();
    {   // gather K (async) / V (manual transpose) chunk rows
      int i = tid >> 1, ch = (tid & 1) * 32;
      int grow = b * LVL_ROWS + gw[s0 + i];
      const __bf16* ksrc = kb + (size_t)grow * Cc + h * HDd + ch;
#pragma unroll
      for (int j = 0; j < 4; ++j) lds_cp8(&kT[i][ch + j * 8], ksrc + j * 8);
      const __bf16* vsrc = vb + (size_t)grow * Cc + h * HDd + ch;
      v8bf vv[4];
#pragma unroll
      for (int j = 0; j < 4; ++j) vv[j] = *(const v8bf*)(vsrc + j * 8);
#pragma unroll
      for (int j = 0; j < 32; ++j) vT[ch + j][i] = vv[j >> 3][j & 7];
    }
    async_wait();
    __syncthreads();

    // logits = q @ K^T  (M=q, N=s, K=d=64 -> two 32-K steps)
    v8f lg[4];
#pragma unroll
    for (int i = 0; i < 4; ++i)
#pragma unroll
      for (int j = 0; j < 8; ++j) lg[i][j] = 0.f;
#pragma unroll
    for (int kk = 0; kk < 2; ++kk) {
      v16bf af = make_frag(&qT[m0 + l16][kk * 32 + lhi * 8],
                           &qT[m0 + l16][kk * 32 + 16 + lhi * 8]);
      v16bf bf[4];
#pragma unroll
      for (int nf = 0; nf < 4; ++nf)
        bf[nf] = make_frag(&kT[nf * 16 + l16][kk * 32 + lhi * 8],
                           &kT[nf * 16 + l16][kk * 32 + 16 + lhi * 8]);
#pragma unroll
      for (int nf = 0; nf < 4; ++nf)
        lg[nf] = __builtin_amdgcn_wmma_f32_16x16x32_bf16(
            false, af, false, bf[nf], (short)0, lg[nf], false, false);
    }
    // + relative position bias
#pragma unroll
    for (int nf = 0; nf < 4; ++nf) {
      int s = s0 + nf * 16 + l16;
#pragma unroll
      for (int r = 0; r < 8; ++r) {
        int qrow = m0 + lhi * 8 + r;
        size_t bi = (((size_t)(w * WSQ + qrow)) * Sdim + s) * Hh + h;
        lg[nf][r] += (float)biasb[bi];
      }
    }
    // online softmax (rows live across the 16 lanes of each half-group)
#pragma unroll
    for (int r = 0; r < 8; ++r) {
      float mx = fmaxf(fmaxf(lg[0][r], lg[1][r]), fmaxf(lg[2][r], lg[3][r]));
#pragma unroll
      for (int off = 1; off < 16; off <<= 1) mx = fmaxf(mx, __shfl_xor(mx, off, 32));
      float mnew  = fmaxf(m_run[r], mx);
      float alpha = __expf(m_run[r] - mnew);
      m_run[r] = mnew;
      float psum = 0.f;
#pragma unroll
      for (int nf = 0; nf < 4; ++nf) {
        float p = __expf(lg[nf][r] - mnew);
        pT[wv][lhi * 8 + r][nf * 16 + l16] = (__bf16)p;
        psum += p;
      }
#pragma unroll
      for (int off = 1; off < 16; off <<= 1) psum += __shfl_xor(psum, off, 32);
      l_run[r] = l_run[r] * alpha + psum;
#pragma unroll
      for (int nf = 0; nf < 4; ++nf) o[nf][r] *= alpha;
    }
    // O += P @ V   (M=q, N=d, K=s=64)
#pragma unroll
    for (int kk = 0; kk < 2; ++kk) {
      v16bf ap = make_frag(&pT[wv][l16][kk * 32 + lhi * 8],
                           &pT[wv][l16][kk * 32 + 16 + lhi * 8]);
      v16bf bv[4];
#pragma unroll
      for (int nf = 0; nf < 4; ++nf)
        bv[nf] = make_frag(&vT[nf * 16 + l16][kk * 32 + lhi * 8],
                           &vT[nf * 16 + l16][kk * 32 + 16 + lhi * 8]);
#pragma unroll
      for (int nf = 0; nf < 4; ++nf)
        o[nf] = __builtin_amdgcn_wmma_f32_16x16x32_bf16(
            false, ap, false, bv[nf], (short)0, o[nf], false, false);
    }
  }

  // normalize and write
#pragma unroll
  for (int nf = 0; nf < 4; ++nf) {
    int d = nf * 16 + l16;
#pragma unroll
    for (int r = 0; r < 8; ++r) {
      int qrow = m0 + lhi * 8 + r;
      float v = o[nf][r] / l_run[r];
      ao[((size_t)(b * Lm + w * WSQ + qrow)) * Cc + h * HDd + d] = (__bf16)v;
    }
  }
}

// ---------------------------------------------------------------- host
extern "C" void kernel_launch(void* const* d_in, const int* in_sizes, int n_in,
                              void* d_out, int out_size, void* d_ws, size_t ws_size,
                              hipStream_t stream) {
  (void)in_sizes; (void)n_in; (void)out_size; (void)ws_size;
  const float* x      = (const float*)d_in[0];
  const float* ln1_s  = (const float*)d_in[1];
  const float* ln1_b  = (const float*)d_in[2];
  const float* wq     = (const float*)d_in[3];
  const float* bq     = (const float*)d_in[4];
  const float* wkv    = (const float*)d_in[5];
  const float* bkv    = (const float*)d_in[6];
  const float* wproj  = (const float*)d_in[7];
  const float* bproj  = (const float*)d_in[8];
  const float* rpb_w1 = (const float*)d_in[9];
  const float* rpb_b1 = (const float*)d_in[10];
  const float* rpb_w2 = (const float*)d_in[11];
  const float* rpb_b2 = (const float*)d_in[12];
  const float* ln2_s  = (const float*)d_in[13];
  const float* ln2_b  = (const float*)d_in[14];
  const float* mlp_w1 = (const float*)d_in[15];
  const float* mlp_b1 = (const float*)d_in[16];
  const float* mlp_w2 = (const float*)d_in[17];
  const float* mlp_b2 = (const float*)d_in[18];
  float* outf = (float*)d_out;

  char* ws = (char*)d_ws;
  size_t off = 0;
  auto alloc = [&](size_t bytes) -> void* {
    void* p = ws + off;
    off = (off + bytes + 255) & ~(size_t)255;
    return p;
  };
  __bf16* wq_b   = (__bf16*)alloc((size_t)512 * 512  * 2);
  __bf16* wkv_b  = (__bf16*)alloc((size_t)512 * 1024 * 2);
  __bf16* wpr_b  = (__bf16*)alloc((size_t)512 * 512  * 2);
  __bf16* w1_b   = (__bf16*)alloc((size_t)512 * 2048 * 2);
  __bf16* w2_b   = (__bf16*)alloc((size_t)2048 * 512 * 2);
  __bf16* hb     = (__bf16*)alloc((size_t)16384 * 512 * 2);
  __bf16* p2b    = (__bf16*)alloc((size_t)8192  * 512 * 2);
  __bf16* p4b    = (__bf16*)alloc((size_t)4096  * 512 * 2);
  __bf16* qb     = (__bf16*)alloc((size_t)16384 * 512 * 2);
  __bf16* kb     = (__bf16*)alloc((size_t)28672 * 512 * 2);
  __bf16* vb     = (__bf16*)alloc((size_t)28672 * 512 * 2);
  __bf16* biasb  = (__bf16*)alloc((size_t)64 * 64 * 768 * 8 * 2);
  int*    gidx   = (int*)   alloc((size_t)64 * 768 * 4);
  __bf16* ao     = (__bf16*)alloc((size_t)16384 * 512 * 2);
  __bf16* h2b    = (__bf16*)alloc((size_t)16384 * 512 * 2);
  __bf16* midb   = (__bf16*)alloc((size_t)16384 * 2048 * 2);

  auto cvt = [&](const float* s, __bf16* d, int n) {
    f32_to_bf16_kernel<<<(n + 255) / 256, 256, 0, stream>>>(s, d, n);
  };
  cvt(wq,     wq_b,  512 * 512);
  cvt(wkv,    wkv_b, 512 * 1024);
  cvt(wproj,  wpr_b, 512 * 512);
  cvt(mlp_w1, w1_b,  512 * 2048);
  cvt(mlp_w2, w2_b,  2048 * 512);

  // LN1 + pooled key levels
  ln_bf16_kernel<<<2048, 256, 0, stream>>>(x, ln1_s, ln1_b, hb, 16384);
  pool_kernel<<<(8192 * 512 + 255) / 256, 256, 0, stream>>>(hb, p2b, 2, 8192);
  pool_kernel<<<(4096 * 512 + 255) / 256, 256, 0, stream>>>(hb, p4b, 4, 4096);

  // Q = h @ wq + bq (scaled)
  gemm_wmma_kernel<<<dim3(4, 128), 256, 0, stream>>>(
      hb, 512, wq_b, 512, 512, bq, MODE_Q, qb, nullptr, nullptr, nullptr, 0, 0);
  // KV per focal level
  gemm_wmma_kernel<<<dim3(8, 128), 256, 0, stream>>>(
      hb, 512, wkv_b, 1024, 512, bkv, MODE_KV, kb, nullptr, nullptr, vb, 4096, 0);
  gemm_wmma_kernel<<<dim3(8, 64), 256, 0, stream>>>(
      p2b, 512, wkv_b, 1024, 512, bkv, MODE_KV, kb, nullptr, nullptr, vb, 2048, 4096);
  gemm_wmma_kernel<<<dim3(8, 32), 256, 0, stream>>>(
      p4b, 512, wkv_b, 1024, 512, bkv, MODE_KV, kb, nullptr, nullptr, vb, 1024, 6144);

  // static gather table + relative position bias
  gidx_kernel<<<(64 * 768 + 255) / 256, 256, 0, stream>>>(gidx);
  rpb_kernel<<<dim3(64, 64), 256, 0, stream>>>(rpb_w1, rpb_b1, rpb_w2, rpb_b2, biasb);

  // flash attention over gathered multi-scale keys
  flash_attn_kernel<<<dim3(8, 64, 4), 128, 0, stream>>>(qb, kb, vb, biasb, gidx, ao);

  // x2 = shortcut + attn_out @ wproj + bproj   (fp32 into d_out)
  gemm_wmma_kernel<<<dim3(4, 128), 256, 0, stream>>>(
      ao, 512, wpr_b, 512, 512, bproj, MODE_PROJ, nullptr, outf, x, nullptr, 0, 0);

  // MLP branch
  ln_bf16_kernel<<<2048, 256, 0, stream>>>(outf, ln2_s, ln2_b, h2b, 16384);
  gemm_wmma_kernel<<<dim3(16, 128), 256, 0, stream>>>(
      h2b, 512, w1_b, 2048, 512, mlp_b1, MODE_MLP1, midb, nullptr, nullptr, nullptr, 0, 0);
  gemm_wmma_kernel<<<dim3(4, 128), 256, 0, stream>>>(
      midb, 2048, w2_b, 512, 2048, mlp_b2, MODE_MLP2, nullptr, outf, outf, nullptr, 0, 0);
}
